// RNNTLoss_91336774517040
// MI455X (gfx1250) — compile-verified, matrix operand order checked by reading, not run on hardware
//
#include <hip/hip_runtime.h>
#include <hip/hip_bf16.h>
#include <math.h>

// Problem constants (from reference): B=16, T=200, U=100, U1=101, V=512
#define RB   16
#define RT   200
#define RU1  101
#define RV   512
#define ROWS (RB * RT * RU1)      // 323200 softmax rows
#define PERB (RT * RU1)           // 20200 lattice cells per utterance
#define NEGV (-1e30f)

typedef __attribute__((ext_vector_type(16))) __bf16 v16bf;
typedef __attribute__((ext_vector_type(8)))  float  v8f;

// ---------------------------------------------------------------------------
// Kernel 1: streaming log-softmax. One wave32 handles a tile of 16 rows.
// Lane L: p = L&15 -> row (tileBase+p), h = L>>4 -> which half of the row.
// Pass 1: row max in f32 VALU. Pass 2: exp(x-max) -> bf16 -> WMMA with an
// all-ones B matrix accumulating the 512-wide row sums into f32 C/D.
// Epilogue: D[m][0] gathered via shfl, blank/emit log-probs written to ws.
// ---------------------------------------------------------------------------
__global__ __launch_bounds__(256) void rnnt_softmax_kernel(
    const float* __restrict__ acts,
    const int*   __restrict__ labels,
    const int*   __restrict__ label_lens,
    float* __restrict__ blank_lp,
    float* __restrict__ emit_lp)
{
    const int lane = threadIdx.x & 31;
    const int wave = threadIdx.x >> 5;
    const int p    = lane & 15;   // row within tile
    const int h    = lane >> 4;   // half of row (0 or 1)

    const long tile = (long)blockIdx.x * 8 + wave;   // 20200 tiles total
    const long R    = tile * 16;                     // first row of tile
    const long rowBase = (R + p) * (long)RV;

    // ---- pass 1: per-row max (each lane scans 256 contiguous floats) ----
    const float4* rp4 = (const float4*)(acts + rowBase + h * 256);
    float mx = -INFINITY;
#pragma unroll 8
    for (int i = 0; i < 64; ++i) {
        float4 v = rp4[i];
        mx = fmaxf(mx, fmaxf(fmaxf(v.x, v.y), fmaxf(v.z, v.w)));
    }
    // combine the two half-row lanes: both lanes of the pair get the row max
    mx = fmaxf(mx, __shfl_xor(mx, 16, 32));

    // ---- pass 2: WMMA-accumulated sum of exp(x - mx), re-read is L2-hot ----
    v16bf bones;
#pragma unroll
    for (int i = 0; i < 16; ++i) bones[i] = (__bf16)1.0f;

    v8f acc = {};   // f32 accumulator, D[m][n] = row-m sum (all n identical)

#pragma unroll
    for (int c = 0; c < 16; ++c) {
        const float4* q = (const float4*)(acts + rowBase + c * 32 + h * 16);
        float4 v0 = q[0], v1 = q[1], v2 = q[2], v3 = q[3];
        v16bf a;
#define EXPB(x) ((__bf16)__expf((x) - mx))
        a[0]  = EXPB(v0.x); a[1]  = EXPB(v0.y); a[2]  = EXPB(v0.z); a[3]  = EXPB(v0.w);
        a[4]  = EXPB(v1.x); a[5]  = EXPB(v1.y); a[6]  = EXPB(v1.z); a[7]  = EXPB(v1.w);
        a[8]  = EXPB(v2.x); a[9]  = EXPB(v2.y); a[10] = EXPB(v2.z); a[11] = EXPB(v2.w);
        a[12] = EXPB(v3.x); a[13] = EXPB(v3.y); a[14] = EXPB(v3.z); a[15] = EXPB(v3.w);
#undef EXPB
        acc = __builtin_amdgcn_wmma_f32_16x16x32_bf16(
            /*neg_a=*/false, a, /*neg_b=*/false, bones,
            /*c_mod=*/(short)0, acc, /*reuse_a=*/false, /*reuse_b=*/false);
    }

    // ---- gather my row's sum: D[m][0]; m<8 lives on lanes 0-15 (VGPR m),
    //      m>=8 on lanes 16-31 (VGPR m-8). Broadcast column 0. ----
    float mysum = 0.0f;
#pragma unroll
    for (int j = 0; j < 8; ++j) {
        float lo = __shfl(acc[j], 0, 32);    // D[j][0]
        float hi = __shfl(acc[j], 16, 32);   // D[j+8][0]
        if (p == j)     mysum = lo;
        if (p == j + 8) mysum = hi;
    }

    // ---- epilogue: lanes 0-15 each finalize one row ----
    if (lane < 16) {
        const long r = R + lane;
        const int  u = (int)(r % RU1);
        const int  b = (int)(r / PERB);
        const float lse = mx + __logf(mysum);
        const int lab = (u < (RU1 - 1)) ? labels[b * (RU1 - 1) + u] : 0;
        const float a0 = acts[r * (long)RV];
        const float al = acts[r * (long)RV + lab];
        blank_lp[r] = a0 - lse;
        emit_lp[r]  = (u < label_lens[b]) ? (al - lse) : NEGV;
    }
}

// ---------------------------------------------------------------------------
// Kernel 2: forward DP over the (t,u) lattice, anti-diagonal wavefront.
// One workgroup per utterance; alpha + blank + emit all staged in LDS
// (3 * 80.8 KB = 242.4 KB of the WGP's 320 KB). The blank/emit preload is
// done with the gfx1250 async global->LDS pipe (no VGPR staging), completed
// with s_wait_asynccnt before the workgroup barrier.
// alpha[t][u] = logaddexp(alpha[t-1][u] + blank[t-1][u],
//                         alpha[t][u-1] +  emit[t][u-1])
// ---------------------------------------------------------------------------
__device__ __forceinline__ float lae(float x, float y) {
    float m = fmaxf(x, y);
    float n = fminf(x, y);
    return m + __logf(1.0f + __expf(n - m));
}

// Async copy of 16 bytes global -> LDS. Generic pointers to __shared__ carry
// the LDS byte offset in their low 32 bits (ISA 10.2),
// which is exactly what the VDST operand of the async op wants.
__device__ __forceinline__ void async_g2l_b128(void* lds_dst, const void* gsrc) {
    unsigned           ldsa = (unsigned)(unsigned long long)lds_dst;
    unsigned long long ga   = (unsigned long long)gsrc;
    asm volatile("global_load_async_to_lds_b128 %0, %1, off"
                 :: "v"(ldsa), "v"(ga) : "memory");
}

__global__ __launch_bounds__(128) void rnnt_dp_kernel(
    const float* __restrict__ blank_lp,
    const float* __restrict__ emit_lp,
    const int*   __restrict__ act_lens,
    const int*   __restrict__ label_lens,
    float* __restrict__ costs)
{
    const int b   = blockIdx.x;
    const int tid = threadIdx.x;

    extern __shared__ float smem[];
    float* alpha = smem;                 // PERB floats
    float* sb    = smem + PERB;          // blank log-probs
    float* se    = smem + 2 * PERB;      // emit log-probs

    const long base = (long)b * PERB;
    // 80800 bytes per array = 5050 exact b128 transfers, all 16B-aligned.
    const int nXfer = (PERB * 4) / 16;   // 5050
    for (int i = tid; i < nXfer; i += 128) {
        async_g2l_b128(&sb[i * 4], &blank_lp[base + i * 4]);
        async_g2l_b128(&se[i * 4], &emit_lp [base + i * 4]);
    }
    asm volatile("s_wait_asynccnt 0" ::: "memory");
    __syncthreads();

    // diagonals d = t + u, d in [0, T + U1 - 2]
    for (int d = 0; d <= RT + RU1 - 2; ++d) {
        const int tlo = (d > RU1 - 1) ? d - (RU1 - 1) : 0;
        const int thi = (d < RT - 1) ? d : RT - 1;
        const int t   = tlo + tid;
        if (t <= thi) {
            const int u = d - t;
            float v;
            if (t == 0) {
                v = (u == 0) ? 0.0f : alpha[u - 1] + se[u - 1];
            } else if (u == 0) {
                v = alpha[(t - 1) * RU1] + sb[(t - 1) * RU1];
            } else {
                v = lae(alpha[(t - 1) * RU1 + u] + sb[(t - 1) * RU1 + u],
                        alpha[t * RU1 + u - 1]  + se[t * RU1 + u - 1]);
            }
            alpha[t * RU1 + u] = v;
        }
        __syncthreads();
    }

    if (tid == 0) {
        const int tt = act_lens[b] - 1;
        const int uu = label_lens[b];
        costs[b] = -(alpha[tt * RU1 + uu] + sb[tt * RU1 + uu]);
    }
}

// ---------------------------------------------------------------------------
// Kernel 3: scalar mean of the per-utterance costs.
// ---------------------------------------------------------------------------
__global__ void rnnt_final_kernel(const float* __restrict__ costs,
                                  float* __restrict__ out)
{
    if (threadIdx.x == 0 && blockIdx.x == 0) {
        float s = 0.0f;
#pragma unroll
        for (int i = 0; i < RB; ++i) s += costs[i];
        out[0] = s / (float)RB;
    }
}

// ---------------------------------------------------------------------------
extern "C" void kernel_launch(void* const* d_in, const int* in_sizes, int n_in,
                              void* d_out, int out_size, void* d_ws, size_t ws_size,
                              hipStream_t stream)
{
    const float* acts       = (const float*)d_in[0];
    const int*   labels     = (const int*)  d_in[1];
    const int*   act_lens   = (const int*)  d_in[2];
    const int*   label_lens = (const int*)  d_in[3];
    float*       out        = (float*)d_out;

    float* blank = (float*)d_ws;        // ROWS floats
    float* emit  = blank + ROWS;        // ROWS floats
    float* costs = emit + ROWS;         // RB floats  (~2.6 MB total in ws)

    // Kernel 1: 20200 wave-tiles of 16 rows, 8 waves per 256-thread block.
    const int nTiles  = ROWS / 16;      // 20200, exact
    const int nBlocks = nTiles / 8;     // 2525, exact
    rnnt_softmax_kernel<<<nBlocks, 256, 0, stream>>>(
        acts, labels, label_lens, blank, emit);

    // Kernel 2: one workgroup per utterance, 242.4 KB dynamic LDS.
    const size_t ldsBytes = (size_t)3 * PERB * sizeof(float);
    rnnt_dp_kernel<<<RB, 128, ldsBytes, stream>>>(
        blank, emit, act_lens, label_lens, costs);

    // Kernel 3: mean.
    rnnt_final_kernel<<<1, 32, 0, stream>>>(costs, out);
}
